// Equiformer_14310831031030
// MI455X (gfx1250) — compile-verified, matrix operand order checked by reading, not run on hardware
//
#include <hip/hip_runtime.h>
#include <hip/hip_bf16.h>
#include <cstdint>

#define N_NODES 8192
#define N_EDGES 32768
#define HEADS   8
#define MDIM    8

typedef __attribute__((ext_vector_type(16))) _Float16 v16h;
typedef __attribute__((ext_vector_type(8)))  _Float16 v8h;
typedef __attribute__((ext_vector_type(8)))  float    v8f;

__device__ __forceinline__ float sigf(float x) { return 1.f / (1.f + __expf(-x)); }
__device__ __forceinline__ float siluf(float x) { return x * sigf(x); }

// Async global->LDS copy (CDNA5): per-lane 4B, tracked by ASYNCcnt.
// LDS byte address = low 32 bits of generic pointer (ISA 10.2 LDS aperture).
__device__ __forceinline__ void async_copy_f32(uint32_t lds_addr, const float* gptr) {
    asm volatile("global_load_async_to_lds_b32 %0, %1, off"
                 :: "v"(lds_addr), "v"((uint64_t)(uintptr_t)gptr)
                 : "memory");
}
__device__ __forceinline__ void async_wait0() {
    asm volatile("s_wait_asynccnt 0x0" ::: "memory");
}

// ---------------------------------------------------------------------------
// Generic WMMA GEMM:  C[R,O] = A[R,K](f16) @ Wh[O,K](f16)^T + bias
// R mult of 16, O mult of 32, K mult of 32. One wave32 owns a 16x32 output
// slab (two 16x16 WMMA tiles sharing the A fragment).
// ---------------------------------------------------------------------------
__global__ __launch_bounds__(256) void gemm_f16_wmma(
    const _Float16* __restrict__ A, const _Float16* __restrict__ Wh,
    const float* __restrict__ bias, float* __restrict__ C,
    int R, int K, int O)
{
    const int lane = threadIdx.x & 31;
    const int wv   = threadIdx.x >> 5;
    const int tilesP = O >> 5;                       // pairs of n-tiles
    const int tilesM = R >> 4;
    const int tile = blockIdx.x * 8 + wv;            // wave-uniform
    if (tile >= tilesM * tilesP) return;             // whole wave exits together
    const int tm = tile / tilesP;
    const int tp = tile - tm * tilesP;
    const int l15 = lane & 15;
    const int kc  = (lane >> 4) << 3;                // 0 or 8

    const _Float16* arow  = A  + (size_t)(tm * 16 + l15) * K + kc;
    const _Float16* brow0 = Wh + (size_t)(tp * 32 + l15) * K + kc;
    const _Float16* brow1 = brow0 + (size_t)16 * K;

    v8f acc0 = {}, acc1 = {};
    for (int k0 = 0; k0 < K; k0 += 32) {
        __builtin_prefetch(arow + k0 + 128, 0, 0);
        v16h a, b0, b1;
        v8h t0 = *(const v8h*)(arow + k0);
        v8h t1 = *(const v8h*)(arow + k0 + 16);
        v8h u0 = *(const v8h*)(brow0 + k0);
        v8h u1 = *(const v8h*)(brow0 + k0 + 16);
        v8h w0 = *(const v8h*)(brow1 + k0);
        v8h w1 = *(const v8h*)(brow1 + k0 + 16);
#pragma unroll
        for (int i = 0; i < 8; ++i) {
            a[i]  = t0[i]; a[8 + i]  = t1[i];
            b0[i] = u0[i]; b0[8 + i] = u1[i];
            b1[i] = w0[i]; b1[8 + i] = w1[i];
        }
        acc0 = __builtin_amdgcn_wmma_f32_16x16x32_f16(
            false, a, false, b0, (short)0, acc0, false, false);
        acc1 = __builtin_amdgcn_wmma_f32_16x16x32_f16(
            false, a, false, b1, (short)0, acc1, false, false);
    }
    const int col0 = tp * 32 + l15;
    const int col1 = col0 + 16;
    const int row0 = tm * 16 + ((lane >> 4) << 3);
    const float bv0 = bias ? bias[col0] : 0.f;
    const float bv1 = bias ? bias[col1] : 0.f;
#pragma unroll
    for (int j = 0; j < 8; ++j) {
        C[(size_t)(row0 + j) * O + col0] = acc0[j] + bv0;
        C[(size_t)(row0 + j) * O + col1] = acc1[j] + bv1;
    }
}

// ---------------------------------------------------------------------------
// Small utility kernels
// ---------------------------------------------------------------------------
__global__ void fill_f32(float* p, float v, int n) {
    int i = blockIdx.x * blockDim.x + threadIdx.x;
    if (i < n) p[i] = v;
}
__global__ void f32_to_f16(const float* __restrict__ in, _Float16* __restrict__ out, int n) {
    int i = blockIdx.x * blockDim.x + threadIdx.x;
    if (i < n) out[i] = (_Float16)in[i];
}
__global__ void silu_f32_to_f16(const float* __restrict__ in, _Float16* __restrict__ out, int n) {
    int i = blockIdx.x * blockDim.x + threadIdx.x;
    if (i < n) out[i] = (_Float16)siluf(in[i]);
}
// dst[o*Kp + k] = (k<K) ? src[k*O + o] : 0   (transpose + zero-pad, emit f16)
__global__ void transpose_pad_f16(const float* __restrict__ src, _Float16* __restrict__ dst,
                                  int K, int O, int Kp) {
    int i = blockIdx.x * blockDim.x + threadIdx.x;
    if (i >= O * Kp) return;
    int o = i / Kp, k = i - o * Kp;
    dst[i] = (_Float16)((k < K) ? src[k * O + o] : 0.f);
}
// radial-MLP input: [E,64] f16 : col0 = r_ij, cols1..32 = edges, rest 0
__global__ void rad_x_build(const float* __restrict__ r_ij, const float* __restrict__ edges,
                            _Float16* __restrict__ xh, int E) {
    int i = blockIdx.x * blockDim.x + threadIdx.x;
    if (i >= E * 64) return;
    int e = i >> 6, c = i & 63;
    float v = (c == 0) ? r_ij[e] : ((c <= 32) ? edges[e * 32 + (c - 1)] : 0.f);
    xh[i] = (_Float16)v;
}

// ---------------------------------------------------------------------------
// LayerNorm (scalar + vector channels). One wave32 per node; lane owns 2 chans.
// ---------------------------------------------------------------------------
__device__ __forceinline__ float wave_sum(float x) {
#pragma unroll
    for (int o = 16; o > 0; o >>= 1) x += __shfl_xor(x, o, 32);
    return x;
}

__global__ __launch_bounds__(256) void ln_kernel(
    const float* __restrict__ s, const float* __restrict__ v,
    const float* __restrict__ gs, const float* __restrict__ bs,
    const float* __restrict__ gv,
    _Float16* __restrict__ sn_h, _Float16* __restrict__ vn_h, int N)
{
    const int lane = threadIdx.x & 31;
    const int n = blockIdx.x * 8 + (threadIdx.x >> 5);
    if (n >= N) return;
    const int c0 = lane, c1 = lane + 32;
    float a = s[(size_t)n * 64 + c0], b = s[(size_t)n * 64 + c1];
    float mu = wave_sum(a + b) * (1.f / 64.f);
    float d0 = a - mu, d1 = b - mu;
    float var = wave_sum(d0 * d0 + d1 * d1) * (1.f / 64.f);
    float inv = rsqrtf(var + 1e-5f);
    sn_h[(size_t)n * 64 + c0] = (_Float16)(d0 * inv * gs[c0] + bs[c0]);
    sn_h[(size_t)n * 64 + c1] = (_Float16)(d1 * inv * gs[c1] + bs[c1]);

    const float* vp0 = v + ((size_t)n * 64 + c0) * 3;
    const float* vp1 = v + ((size_t)n * 64 + c1) * 3;
    float q = vp0[0]*vp0[0] + vp0[1]*vp0[1] + vp0[2]*vp0[2]
            + vp1[0]*vp1[0] + vp1[1]*vp1[1] + vp1[2]*vp1[2];
    float n2 = wave_sum(q) * (1.f / 64.f);
    float invv = rsqrtf(n2 + 1e-5f);
#pragma unroll
    for (int k = 0; k < 3; ++k) {
        vn_h[((size_t)n * 3 + k) * 64 + c0] = (_Float16)(vp0[k] * invv * gv[c0]);
        vn_h[((size_t)n * 3 + k) * 64 + c1] = (_Float16)(vp1[k] * invv * gv[c1]);
    }
}

__global__ __launch_bounds__(256) void resid_ln_kernel(
    const float* __restrict__ s0, const float* __restrict__ v0,
    const float* __restrict__ ds, const float* __restrict__ dvt,
    const float* __restrict__ gs, const float* __restrict__ bs,
    const float* __restrict__ gv,
    float* __restrict__ s1, float* __restrict__ v1t,
    _Float16* __restrict__ sn_h, _Float16* __restrict__ vn_h, int N)
{
    const int lane = threadIdx.x & 31;
    const int n = blockIdx.x * 8 + (threadIdx.x >> 5);
    if (n >= N) return;
    const int c0 = lane, c1 = lane + 32;
    float a = s0[(size_t)n * 64 + c0] + ds[(size_t)n * 64 + c0];
    float b = s0[(size_t)n * 64 + c1] + ds[(size_t)n * 64 + c1];
    s1[(size_t)n * 64 + c0] = a;
    s1[(size_t)n * 64 + c1] = b;
    float mu = wave_sum(a + b) * (1.f / 64.f);
    float d0 = a - mu, d1 = b - mu;
    float var = wave_sum(d0 * d0 + d1 * d1) * (1.f / 64.f);
    float inv = rsqrtf(var + 1e-5f);
    sn_h[(size_t)n * 64 + c0] = (_Float16)(d0 * inv * gs[c0] + bs[c0]);
    sn_h[(size_t)n * 64 + c1] = (_Float16)(d1 * inv * gs[c1] + bs[c1]);

    float vb0[3], vb1[3], q = 0.f;
#pragma unroll
    for (int k = 0; k < 3; ++k) {
        vb0[k] = v0[((size_t)n * 64 + c0) * 3 + k] + dvt[((size_t)n * 3 + k) * 64 + c0];
        vb1[k] = v0[((size_t)n * 64 + c1) * 3 + k] + dvt[((size_t)n * 3 + k) * 64 + c1];
        v1t[((size_t)n * 3 + k) * 64 + c0] = vb0[k];
        v1t[((size_t)n * 3 + k) * 64 + c1] = vb1[k];
        q += vb0[k] * vb0[k] + vb1[k] * vb1[k];
    }
    float n2 = wave_sum(q) * (1.f / 64.f);
    float invv = rsqrtf(n2 + 1e-5f);
#pragma unroll
    for (int k = 0; k < 3; ++k) {
        vn_h[((size_t)n * 3 + k) * 64 + c0] = (_Float16)(vb0[k] * invv * gv[c0]);
        vn_h[((size_t)n * 3 + k) * 64 + c1] = (_Float16)(vb1[k] * invv * gv[c1]);
    }
}

// ---------------------------------------------------------------------------
// Edge tensor-product chain. One thread per (edge, head). Per-head weights
// staged into LDS with GLOBAL_LOAD_ASYNC_TO_LDS (ASYNCcnt) instead of the
// VGPR round-trip; compute fused per mid-channel.
// ---------------------------------------------------------------------------
__global__ __launch_bounds__(256) void edge_chain_kernel(
    const float* __restrict__ s_dstL, const float* __restrict__ v_dstL,
    const float* __restrict__ s_srcL, const float* __restrict__ v_srcL,
    const int* __restrict__ srcIdx, const int* __restrict__ dstIdx,
    const float* __restrict__ rvec, const float* __restrict__ wcut,
    const float* __restrict__ radw,
    const float* __restrict__ w_s_init, const float* __restrict__ b_s_init,
    const float* __restrict__ w_v_init,
    const float* __restrict__ dtp_w_s, const float* __restrict__ dtp_b_s,
    const float* __restrict__ dtp_w_v,
    const float* __restrict__ am_w_s, const float* __restrict__ am_b_s,
    const float* __restrict__ am_w_v, const float* __restrict__ attn_w,
    float* __restrict__ msg_s, float* __restrict__ msg_v,
    float* __restrict__ zbuf, float* __restrict__ zmax, int E)
{
    __shared__ float smem[5432];
    float* lw_s    = smem;            // 2048
    float* lw_v    = smem + 2048;     // 2048
    float* ldtp_ws = smem + 4096;     // 768
    float* ldtp_wv = smem + 4864;     // 256
    float* lam_ws  = smem + 5120;     // 128
    float* lam_wv  = smem + 5248;     // 128
    float* lb_s    = smem + 5376;     // 16
    float* ldtp_bs = smem + 5392;     // 24
    float* lam_bs  = smem + 5416;     // 8
    float* lattn_w = smem + 5424;     // 8

    const int h = blockIdx.y;
    const int tid = threadIdx.x;
    const uint32_t sb = (uint32_t)(uintptr_t)smem;   // LDS byte base
    for (int i = tid; i < 2048; i += 256) {
        async_copy_f32(sb + 4u * i,           w_s_init + h * 2048 + i);
        async_copy_f32(sb + 4u * (2048 + i),  w_v_init + h * 2048 + i);
    }
    for (int i = tid; i < 768; i += 256)
        async_copy_f32(sb + 4u * (4096 + i), dtp_w_s + h * 768 + i);
    async_copy_f32(sb + 4u * (4864 + tid), dtp_w_v + h * 256 + tid);
    if (tid < 128) {
        async_copy_f32(sb + 4u * (5120 + tid), am_w_s + h * 128 + tid);
        async_copy_f32(sb + 4u * (5248 + tid), am_w_v + h * 128 + tid);
    }
    if (tid < 16) async_copy_f32(sb + 4u * (5376 + tid), b_s_init + h * 16 + tid);
    if (tid < 24) async_copy_f32(sb + 4u * (5392 + tid), dtp_b_s + h * 24 + tid);
    if (tid < 8) {
        async_copy_f32(sb + 4u * (5416 + tid), am_b_s + h * 8 + tid);
        async_copy_f32(sb + 4u * (5424 + tid), attn_w + h * 8 + tid);
    }
    async_wait0();
    __syncthreads();

    const int e = blockIdx.x * blockDim.x + tid;
    if (e >= E) return;
    const int di = dstIdx[e], si = srcIdx[e];
    const int cb = h * MDIM;

    float s_i[8], s_j[8], v_i[8][3], v_j[8][3];
#pragma unroll
    for (int m = 0; m < 8; ++m) {
        s_i[m] = s_dstL[(size_t)di * 64 + cb + m];
        s_j[m] = s_srcL[(size_t)si * 64 + cb + m];
    }
#pragma unroll
    for (int k = 0; k < 3; ++k)
#pragma unroll
        for (int m = 0; m < 8; ++m) {
            v_i[m][k] = v_dstL[((size_t)di * 3 + k) * 64 + cb + m];
            v_j[m][k] = v_srcL[((size_t)si * 3 + k) * 64 + cb + m];
        }
    const float r0 = rvec[e * 3 + 0], r1 = rvec[e * 3 + 1], r2 = rvec[e * 3 + 2];
    const float* wr = radw + (size_t)e * 512;

    float s2[24], v2[8][3];
    for (int o = 0; o < 24; ++o) s2[o] = ldtp_bs[o];
#pragma unroll
    for (int o = 0; o < 8; ++o) { v2[o][0] = 0.f; v2[o][1] = 0.f; v2[o][2] = 0.f; }

    for (int c = 0; c < 16; ++c) {
        float as = lb_s[c], av0 = 0.f, av1 = 0.f, av2 = 0.f;
        const float* ws_c = &lw_s[c * 128];
        const float* wv_c = &lw_v[c * 128];
        for (int i = 0; i < 8; ++i) {
            const float si_ = s_i[i];
            const float vi0 = v_i[i][0], vi1 = v_i[i][1], vi2 = v_i[i][2];
#pragma unroll
            for (int j = 0; j < 8; ++j) {
                const int idx = i * 8 + j;
                const float sj_ = s_j[j];
                const float vj0 = v_j[j][0], vj1 = v_j[j][1], vj2 = v_j[j][2];
                const float w1 = ws_c[idx], w2 = ws_c[64 + idx];
                as += w1 * (si_ * sj_) + w2 * (vi0 * vj0 + vi1 * vj1 + vi2 * vj2);
                const float u1 = wv_c[idx], u2 = wv_c[64 + idx];
                av0 += u1 * (si_ * vj0) + u2 * (vi0 * sj_);
                av1 += u1 * (si_ * vj1) + u2 * (vi1 * sj_);
                av2 += u1 * (si_ * vj2) + u2 * (vi2 * sj_);
            }
        }
        const float w_ss = wr[0 * 128 + h * 16 + c];
        const float w_sv = wr[1 * 128 + h * 16 + c];
        const float w_vs = wr[2 * 128 + h * 16 + c];
        const float w_vv = wr[3 * 128 + h * 16 + c];
        const float sdc  = w_ss * as;
        const float sdc2 = w_vv * (av0 * r0 + av1 * r1 + av2 * r2);
        const float vd00 = w_sv * as * r0, vd01 = w_sv * as * r1, vd02 = w_sv * as * r2;
        const float vd10 = w_vs * av0,     vd11 = w_vs * av1,     vd12 = w_vs * av2;
        for (int o = 0; o < 24; ++o)
            s2[o] += ldtp_ws[o * 32 + c] * sdc + ldtp_ws[o * 32 + 16 + c] * sdc2;
#pragma unroll
        for (int o = 0; o < 8; ++o) {
            const float wa = ldtp_wv[o * 32 + c], wb = ldtp_wv[o * 32 + 16 + c];
            v2[o][0] += wa * vd00 + wb * vd10;
            v2[o][1] += wa * vd01 + wb * vd11;
            v2[o][2] += wa * vd02 + wb * vd12;
        }
    }

    float sact[8], vg[8][3], smv[16];
#pragma unroll
    for (int c = 0; c < 8; ++c) {
        sact[c] = siluf(s2[16 + c]);
        const float g = sigf(s2[8 + c]);
        vg[c][0] = g * v2[c][0]; vg[c][1] = g * v2[c][1]; vg[c][2] = g * v2[c][2];
        smv[c] = sact[c];
        smv[8 + c] = vg[c][0] * r0 + vg[c][1] * r1 + vg[c][2] * r2;
    }

#pragma unroll
    for (int o = 0; o < 8; ++o) {
        float acc = lam_bs[o];
#pragma unroll
        for (int n = 0; n < 16; ++n) acc += lam_ws[o * 16 + n] * smv[n];
        msg_s[(size_t)e * 64 + cb + o] = acc;
        float a0 = 0.f, a1 = 0.f, a2 = 0.f;
#pragma unroll
        for (int n = 0; n < 8; ++n) {
            const float w1 = lam_wv[o * 16 + n];
            const float w2 = lam_wv[o * 16 + 8 + n];
            a0 += w1 * (sact[n] * r0) + w2 * vg[n][0];
            a1 += w1 * (sact[n] * r1) + w2 * vg[n][1];
            a2 += w1 * (sact[n] * r2) + w2 * vg[n][2];
        }
        float* mv = msg_v + ((size_t)e * 64 + cb + o) * 3;
        mv[0] = a0; mv[1] = a1; mv[2] = a2;
    }

    float zr = 0.f;
#pragma unroll
    for (int n = 0; n < 8; ++n) zr += lattn_w[n] * s2[n];
    const float z = wcut[e] * ((zr > 20.f) ? zr : log1pf(__expf(zr)));
    zbuf[(size_t)e * 8 + h] = z;                 // z >= 0 always
    atomicMax((int*)&zmax[(size_t)di * 8 + h], __float_as_int(z));
}

__global__ __launch_bounds__(256) void softmax_pass2(
    const int* __restrict__ dstIdx, float* __restrict__ zbuf,
    const float* __restrict__ zmax, float* __restrict__ denom, int E)
{
    int i = blockIdx.x * blockDim.x + threadIdx.x;
    if (i >= E * 8) return;
    int e = i >> 3, h = i & 7;
    int di = dstIdx[e];
    float ev = __expf(zbuf[i] - zmax[(size_t)di * 8 + h]);
    zbuf[i] = ev;
    atomicAdd(&denom[(size_t)di * 8 + h], ev);
}

__global__ __launch_bounds__(256) void scatter_kernel(
    const int* __restrict__ dstIdx, const float* __restrict__ zbuf,
    const float* __restrict__ denom,
    const float* __restrict__ msg_s, const float* __restrict__ msg_v,
    float* __restrict__ s_agg, float* __restrict__ v_agg, int E)
{
    int i = blockIdx.x * blockDim.x + threadIdx.x;
    if (i >= E * 8) return;
    int e = i >> 3, h = i & 7;
    int di = dstIdx[e];
    float a = zbuf[i] / denom[(size_t)di * 8 + h];
    const int cb = h * MDIM;
#pragma unroll
    for (int m = 0; m < 8; ++m)
        atomicAdd(&s_agg[(size_t)di * 64 + cb + m], a * msg_s[(size_t)e * 64 + cb + m]);
#pragma unroll
    for (int m = 0; m < 8; ++m) {
        const float* mv = msg_v + ((size_t)e * 64 + cb + m) * 3;
#pragma unroll
        for (int k = 0; k < 3; ++k)
            atomicAdd(&v_agg[((size_t)di * 3 + k) * 64 + cb + m], a * mv[k]);
    }
}

__global__ __launch_bounds__(256) void ff_act_kernel(
    const float* __restrict__ sf1, const float* __restrict__ vf1,
    _Float16* __restrict__ s2_h, _Float16* __restrict__ v2_h, int N)
{
    int i = blockIdx.x * blockDim.x + threadIdx.x;
    if (i >= N * 192) return;
    int n = i / 192, c = i - n * 192;
    float x = sf1[(size_t)n * 384 + c];
    float g = sigf(sf1[(size_t)n * 384 + 192 + c]);
    s2_h[i] = (_Float16)siluf(x);
#pragma unroll
    for (int k = 0; k < 3; ++k)
        v2_h[((size_t)n * 3 + k) * 192 + c] =
            (_Float16)(g * vf1[((size_t)n * 3 + k) * 192 + c]);
}

__global__ __launch_bounds__(256) void final_out_kernel(
    const float* __restrict__ s1, const float* __restrict__ sf2,
    const float* __restrict__ v1t, const float* __restrict__ vf2,
    float* __restrict__ out, int N)
{
    int i = blockIdx.x * blockDim.x + threadIdx.x;
    if (i >= N * 64) return;
    int n = i >> 6, c = i & 63;
    out[i] = s1[i] + sf2[i];
    float* ov = out + (size_t)N * 64 + (size_t)i * 3;
#pragma unroll
    for (int k = 0; k < 3; ++k) {
        size_t t = ((size_t)n * 3 + k) * 64 + c;
        ov[k] = v1t[t] + vf2[t];
    }
}

// ---------------------------------------------------------------------------
// Host side
// ---------------------------------------------------------------------------
extern "C" void kernel_launch(void* const* d_in, const int* in_sizes, int n_in,
                              void* d_out, int out_size, void* d_ws, size_t ws_size,
                              hipStream_t stream) {
    (void)in_sizes; (void)n_in; (void)out_size; (void)ws_size;
    const int N = N_NODES, E = N_EDGES;

    const float* s_in   = (const float*)d_in[0];
    const float* v_in   = (const float*)d_in[1];
    const float* edges  = (const float*)d_in[2];
    const float* r_ij   = (const float*)d_in[3];
    const float* r_vec  = (const float*)d_in[4];
    const float* wcut   = (const float*)d_in[5];
    const int*   srcI   = (const int*)d_in[6];
    const int*   dstI   = (const int*)d_in[7];
    auto P = [&](int i) { return (const float*)d_in[8 + i]; };
    const float *ln_a_gs = P(0), *ln_a_bs = P(1), *ln_a_gv = P(2);
    const float *ln_f_gs = P(3), *ln_f_bs = P(4), *ln_f_gv = P(5);
    const float *src_Ws = P(6), *src_bs = P(7), *src_Wv = P(8);
    const float *dst_Ws = P(9), *dst_bs = P(10), *dst_Wv = P(11);
    const float *w_s_init = P(12), *b_s_init = P(13), *w_v_init = P(14);
    const float *rad_W1 = P(15), *rad_b1 = P(16), *rad_W2 = P(17), *rad_b2 = P(18);
    const float *dtp_w_s = P(19), *dtp_b_s = P(20), *dtp_w_v = P(21);
    const float *am_w_s = P(22), *am_b_s = P(23), *am_w_v = P(24), *attn_w = P(25);
    const float *fin_Ws = P(26), *fin_bs = P(27), *fin_Wv = P(28);
    const float *ff1_Ws = P(29), *ff1_bs = P(30), *ff1_Wv = P(31);
    const float *ff2_Ws = P(32), *ff2_bs = P(33), *ff2_Wv = P(34);

    char* wbase = (char*)d_ws;
    size_t off = 0;
    auto alloc = [&](size_t bytes) -> void* {
        void* p = wbase + off;
        off = (off + bytes + 255) & ~(size_t)255;
        return p;
    };
    _Float16* sn_h  = (_Float16*)alloc((size_t)N * 64 * 2);
    _Float16* vn_h  = (_Float16*)alloc((size_t)N * 192 * 2);
    float* s_dst = (float*)alloc((size_t)N * 64 * 4);
    float* v_dst = (float*)alloc((size_t)N * 192 * 4);
    float* s_src = (float*)alloc((size_t)N * 64 * 4);
    float* v_src = (float*)alloc((size_t)N * 192 * 4);
    _Float16* x_h   = (_Float16*)alloc((size_t)E * 64 * 2);
    float* hid  = (float*)alloc((size_t)E * 64 * 4);
    _Float16* hid_h = (_Float16*)alloc((size_t)E * 64 * 2);
    float* radw = (float*)alloc((size_t)E * 512 * 4);
    float* zbuf = (float*)alloc((size_t)E * 8 * 4);
    float* msg_s = (float*)alloc((size_t)E * 64 * 4);
    float* msg_v = (float*)alloc((size_t)E * 192 * 4);
    float* zmax = (float*)alloc((size_t)N * 8 * 4);
    float* denom = (float*)alloc((size_t)N * 8 * 4);
    float* s_agg = (float*)alloc((size_t)N * 64 * 4);
    float* v_agg = (float*)alloc((size_t)N * 192 * 4);
    _Float16* sagg_h = (_Float16*)alloc((size_t)N * 64 * 2);
    _Float16* vagg_h = (_Float16*)alloc((size_t)N * 192 * 2);
    float* s_fin = (float*)alloc((size_t)N * 64 * 4);
    float* v_fin = (float*)alloc((size_t)N * 192 * 4);
    float* s1   = (float*)alloc((size_t)N * 64 * 4);
    float* v1t  = (float*)alloc((size_t)N * 192 * 4);
    _Float16* sn2_h = (_Float16*)alloc((size_t)N * 64 * 2);
    _Float16* vn2_h = (_Float16*)alloc((size_t)N * 192 * 2);
    float* sf1 = (float*)alloc((size_t)N * 384 * 4);
    float* vf1 = (float*)alloc((size_t)N * 576 * 4);
    _Float16* s2_h = (_Float16*)alloc((size_t)N * 192 * 2);
    _Float16* v2_h = (_Float16*)alloc((size_t)N * 576 * 2);
    float* sf2 = (float*)alloc((size_t)N * 64 * 4);
    float* vf2 = (float*)alloc((size_t)N * 192 * 4);
    // f16 weight copies (pre-converted once per call; kills per-tile cvt chains)
    _Float16* dstWs_h = (_Float16*)alloc(64 * 64 * 2);
    _Float16* dstWv_h = (_Float16*)alloc(64 * 64 * 2);
    _Float16* srcWs_h = (_Float16*)alloc(64 * 64 * 2);
    _Float16* srcWv_h = (_Float16*)alloc(64 * 64 * 2);
    _Float16* w1t_h   = (_Float16*)alloc(64 * 64 * 2);
    _Float16* w2t_h   = (_Float16*)alloc(512 * 64 * 2);
    _Float16* finWs_h = (_Float16*)alloc(64 * 64 * 2);
    _Float16* finWv_h = (_Float16*)alloc(64 * 64 * 2);
    _Float16* ff1Ws_h = (_Float16*)alloc(384 * 64 * 2);
    _Float16* ff1Wv_h = (_Float16*)alloc(192 * 64 * 2);
    _Float16* ff2Ws_h = (_Float16*)alloc(64 * 192 * 2);
    _Float16* ff2Wv_h = (_Float16*)alloc(64 * 192 * 2);

    auto gemm = [&](const _Float16* A, const _Float16* Wh, const float* bias,
                    float* C, int R, int K, int O) {
        int tiles = (R / 16) * (O / 32);
        int blocks = (tiles + 7) / 8;
        gemm_f16_wmma<<<blocks, 256, 0, stream>>>(A, Wh, bias, C, R, K, O);
    };
    auto ew = [&](int n) { return dim3((n + 255) / 256); };

    // 0) zero accumulators (deterministic per-call)
    fill_f32<<<ew(N * 64), 256, 0, stream>>>(s_agg, 0.f, N * 64);
    fill_f32<<<ew(N * 192), 256, 0, stream>>>(v_agg, 0.f, N * 192);
    fill_f32<<<ew(N * 8), 256, 0, stream>>>(zmax, 0.f, N * 8);
    fill_f32<<<ew(N * 8), 256, 0, stream>>>(denom, 0.f, N * 8);

    // 1) weight prep: transposes (radial) + f32->f16 copies (all GEMM weights)
    transpose_pad_f16<<<ew(64 * 64), 256, 0, stream>>>(rad_W1, w1t_h, 33, 64, 64);
    transpose_pad_f16<<<ew(512 * 64), 256, 0, stream>>>(rad_W2, w2t_h, 64, 512, 64);
    f32_to_f16<<<ew(64 * 64), 256, 0, stream>>>(dst_Ws, dstWs_h, 64 * 64);
    f32_to_f16<<<ew(64 * 64), 256, 0, stream>>>(dst_Wv, dstWv_h, 64 * 64);
    f32_to_f16<<<ew(64 * 64), 256, 0, stream>>>(src_Ws, srcWs_h, 64 * 64);
    f32_to_f16<<<ew(64 * 64), 256, 0, stream>>>(src_Wv, srcWv_h, 64 * 64);
    f32_to_f16<<<ew(64 * 64), 256, 0, stream>>>(fin_Ws, finWs_h, 64 * 64);
    f32_to_f16<<<ew(64 * 64), 256, 0, stream>>>(fin_Wv, finWv_h, 64 * 64);
    f32_to_f16<<<ew(384 * 64), 256, 0, stream>>>(ff1_Ws, ff1Ws_h, 384 * 64);
    f32_to_f16<<<ew(192 * 64), 256, 0, stream>>>(ff1_Wv, ff1Wv_h, 192 * 64);
    f32_to_f16<<<ew(64 * 192), 256, 0, stream>>>(ff2_Ws, ff2Ws_h, 64 * 192);
    f32_to_f16<<<ew(64 * 192), 256, 0, stream>>>(ff2_Wv, ff2Wv_h, 64 * 192);

    // 2) attention LayerNorm -> f16 operands
    ln_kernel<<<N / 8, 256, 0, stream>>>(s_in, v_in, ln_a_gs, ln_a_bs, ln_a_gv,
                                         sn_h, vn_h, N);

    // 3) dst/src equivariant linears (WMMA)
    gemm(sn_h, dstWs_h, dst_bs, s_dst, N, 64, 64);
    gemm(vn_h, dstWv_h, nullptr, v_dst, 3 * N, 64, 64);
    gemm(sn_h, srcWs_h, src_bs, s_src, N, 64, 64);
    gemm(vn_h, srcWv_h, nullptr, v_src, 3 * N, 64, 64);

    // 4) radial MLP (WMMA)
    rad_x_build<<<ew(E * 64), 256, 0, stream>>>(r_ij, edges, x_h, E);
    gemm(x_h, w1t_h, rad_b1, hid, E, 64, 64);
    silu_f32_to_f16<<<ew(E * 64), 256, 0, stream>>>(hid, hid_h, E * 64);
    gemm(hid_h, w2t_h, rad_b2, radw, E, 64, 512);

    // 5) per-edge tensor-product chain + attention logits
    edge_chain_kernel<<<dim3(E / 256, HEADS), 256, 0, stream>>>(
        s_dst, v_dst, s_src, v_src, srcI, dstI, r_vec, wcut, radw,
        w_s_init, b_s_init, w_v_init, dtp_w_s, dtp_b_s, dtp_w_v,
        am_w_s, am_b_s, am_w_v, attn_w, msg_s, msg_v, zbuf, zmax, E);

    // 6) segment softmax + weighted scatter
    softmax_pass2<<<ew(E * 8), 256, 0, stream>>>(dstI, zbuf, zmax, denom, E);
    scatter_kernel<<<ew(E * 8), 256, 0, stream>>>(dstI, zbuf, denom, msg_s, msg_v,
                                                  s_agg, v_agg, E);

    // 7) final attention linear + residual + FF LayerNorm
    f32_to_f16<<<ew(N * 64), 256, 0, stream>>>(s_agg, sagg_h, N * 64);
    f32_to_f16<<<ew(N * 192), 256, 0, stream>>>(v_agg, vagg_h, N * 192);
    gemm(sagg_h, finWs_h, fin_bs, s_fin, N, 64, 64);
    gemm(vagg_h, finWv_h, nullptr, v_fin, 3 * N, 64, 64);
    resid_ln_kernel<<<N / 8, 256, 0, stream>>>(s_in, v_in, s_fin, v_fin,
                                               ln_f_gs, ln_f_bs, ln_f_gv,
                                               s1, v1t, sn2_h, vn2_h, N);

    // 8) feed-forward block (WMMA)
    gemm(sn2_h, ff1Ws_h, ff1_bs, sf1, N, 64, 384);
    gemm(vn2_h, ff1Wv_h, nullptr, vf1, 3 * N, 64, 192);
    ff_act_kernel<<<ew(N * 192), 256, 0, stream>>>(sf1, vf1, s2_h, v2_h, N);
    gemm(s2_h, ff2Ws_h, ff2_bs, sf2, N, 192, 64);
    gemm(v2_h, ff2Wv_h, nullptr, vf2, 3 * N, 192, 64);

    // 9) final residual -> output layout
    final_out_kernel<<<ew(N * 64), 256, 0, stream>>>(s1, sf2, v1t, vf2,
                                                     (float*)d_out, N);
}
